// RNN_45715631899365
// MI455X (gfx1250) — matrix operand order repaired and checked by reading, not compiled, not run on hardware
//
#include <hip/hip_runtime.h>

// ---------------------------------------------------------------------------
// Persistent 2-layer Elman RNN for MI455X (gfx1250).
//   B=32, T=2048, I=256, H=512, L=2.
//   - 8 persistent workgroups, each owns 64 output columns of both layers.
//   - Transposed bf16 weight slabs (224 KB) resident in LDS for all 2048 steps,
//     staged once via the Tensor Data Mover (tensor_load_to_lds + tensorcnt).
//   - v_wmma_f32_16x16x32_bf16 with fp32 accumulation.
//   - Hidden state exchanged via L2 with a global arrive/spin barrier (2/step).
//   - global_prefetch of x_{t+1}; non-temporal stores for the streamed output.
// ---------------------------------------------------------------------------

#define BB 32
#define TT 2048
#define II 256
#define HH 512
#define NWG 8            // workgroups (N-split of H=512 into slabs of 64)
#define NW  64           // output columns per workgroup
#define THREADS 256      // 8 waves: 2 M-tiles x 4 N-tiles of 16x16

static constexpr size_t SMEM_BYTES = (size_t)NW * (II + 3 * HH) * sizeof(__bf16); // 229376

typedef __attribute__((ext_vector_type(16))) __bf16       v16bf;
typedef __attribute__((ext_vector_type(8)))  float        v8f;
typedef __attribute__((ext_vector_type(4)))  unsigned int u32x4;
typedef __attribute__((ext_vector_type(4)))  unsigned int v4u;
typedef __attribute__((ext_vector_type(8)))  int          v8i;
typedef __attribute__((ext_vector_type(4)))  int          v4i;

#if defined(__has_builtin)
#if __has_builtin(__builtin_amdgcn_tensor_load_to_lds) && \
    __has_builtin(__builtin_amdgcn_s_wait_tensorcnt)
#define HAVE_TDM 1
#endif
#endif
#ifndef HAVE_TDM
#define HAVE_TDM 0
#endif

union FragU { struct { u32x4 lo, hi; } p; v16bf v; };

// A fragment (16x32, M x K, bf16). Lane L: row m = L%16, half = L/16.
// Per ISA: VGPR0-3 hold K = half*8 + [0,8), VGPR4-7 hold K = 16 + half*8 + [0,8).
// rowk points at &A[m][k_base] (16B aligned).
__device__ __forceinline__ v16bf load_fragA(const __bf16* rowk, int half) {
  const u32x4* q = (const u32x4*)rowk;
  FragU u;
  u.p.lo = q[half];
  u.p.hi = q[2 + half];
  return u.v;
}

// B fragment (32x16, K x N, bf16) from N-major (transposed) storage.
// Lane L: col n = L%16, half = L/16; VGPR0-3 hold K = half*8 + [0,8), VGPR4-7 hold K = 16 + half*8 + [0,8).
// rowk points at &Wt[n][k_base] (32B-aligned rows).
__device__ __forceinline__ v16bf load_fragB(const __bf16* rowk, int half) {
  const u32x4* q = (const u32x4*)rowk;
  FragU u;
  u.p.lo = q[half];
  u.p.hi = q[2 + half];
  return u.v;
}

__device__ __forceinline__ float fast_tanh(float x) {
  x = fminf(fmaxf(x, -9.0f), 9.0f);
  float e = __expf(2.0f * x);       // v_exp_f32
  return (e - 1.0f) / (e + 1.0f);   // v_rcp based
}

#define WMMA_BF16(A, Bf, C) \
  __builtin_amdgcn_wmma_f32_16x16x32_bf16(false, (A), false, (Bf), (short)0, (C), false, false)

#if HAVE_TDM
// One 1D TDM descriptor: copy nelem bf16 elements (contiguous) from gsrc to
// LDS byte offset lds_off.  D# bitfields per CDNA5 ISA ch.8:
//   group0: count=1 | lds_addr | global_addr[56:0] | type=2
//   group1: data_size=1(2B), tensor_dim0=tile_dim0=nelem, tensor_dim1=tile_dim1=1,
//           tensor_dim0_stride=nelem
__device__ __forceinline__ void tdm_load_slab(unsigned lds_off, const __bf16* gsrc,
                                              unsigned nelem /* <= 32768 */) {
  unsigned long long ga = (unsigned long long)(uintptr_t)gsrc;
  v4u g0;
  g0[0] = 1u;                                            // count=1, user D#
  g0[1] = lds_off;                                       // lds_addr (bytes)
  g0[2] = (unsigned)(ga & 0xffffffffu);                  // global_addr[31:0]
  g0[3] = (unsigned)((ga >> 32) & 0x01ffffffu) | (2u << 30); // ga[56:32] | type=2
  v8i g1;
  g1[0] = (int)(1u << 16);                               // data_size=1 -> 2 bytes
  g1[1] = (int)((nelem & 0xffffu) << 16);                // tensor_dim0[15:0]
  g1[2] = (int)((nelem >> 16) & 0xffffu) | (1 << 16);    // tensor_dim0[31:16] | tensor_dim1=1
  g1[3] = (int)((nelem & 0xffffu) << 16);                // tile_dim0 = nelem
  g1[4] = 1;                                             // tile_dim1 = 1
  g1[5] = (int)nelem;                                    // tensor_dim0_stride[31:0]
  g1[6] = 0;
  g1[7] = 0;
  v4i gz = {};
#if __clang_major__ >= 23
  v8i gz8 = {};
  __builtin_amdgcn_tensor_load_to_lds(g0, g1, gz, gz, gz8, 0);
#else
  __builtin_amdgcn_tensor_load_to_lds(g0, g1, gz, gz, 0);
#endif
}
#endif

// ---------------------------------------------------------------------------
__global__ void __launch_bounds__(THREADS, 1)
rnn_persistent(const __bf16* __restrict__ xb,    // [B][T][I] bf16
               const __bf16* __restrict__ wt,    // Wih0T[512][256] | Whh0T[512][512] | Wih1T | Whh1T
               const float*  __restrict__ b0g,
               const float*  __restrict__ b1g,
               __bf16*       __restrict__ h0buf, // [2][B][H] double buffered
               __bf16*       __restrict__ h1buf, // [2][B][H]
               float*        __restrict__ out,   // [B][T][H] then hT [2][B][H]
               unsigned*     __restrict__ barcnt)
{
  extern __shared__ __bf16 lds[];
  __bf16* Lih0 = lds;                 // [NW][II]   bytes [0, 32768)
  __bf16* Lhh0 = Lih0 + NW * II;      // [NW][HH]   bytes [32768, 98304)
  __bf16* Lih1 = Lhh0 + NW * HH;      // [NW][HH]   bytes [98304, 163840)
  __bf16* Lhh1 = Lih1 + NW * HH;      // [NW][HH]   bytes [163840, 229376)

  const int tid = threadIdx.x;
  const int wg  = blockIdx.x;
  const int n0  = wg * NW;

  // ---- stage this WG's transposed weight slabs into LDS (once, persistent) ----
  const __bf16* g0s = wt + (size_t)n0 * II;                                         // Wih0T rows n0..
  const __bf16* g1s = wt + (size_t)HH * II + (size_t)n0 * HH;                       // Whh0T
  const __bf16* g2s = wt + (size_t)HH * II + (size_t)HH * HH + (size_t)n0 * HH;     // Wih1T
  const __bf16* g3s = wt + (size_t)HH * II + 2 * (size_t)HH * HH + (size_t)n0 * HH; // Whh1T
#if HAVE_TDM
  if (tid < 32) {   // one wave drives the Tensor Data Mover
    tdm_load_slab(0u,      g0s, NW * II);
    tdm_load_slab(32768u,  g1s, NW * HH);
    tdm_load_slab(98304u,  g2s, NW * HH);
    tdm_load_slab(163840u, g3s, NW * HH);
    __builtin_amdgcn_s_wait_tensorcnt(0);
  }
#else
  {
    const u32x4* s0 = (const u32x4*)g0s; u32x4* d0 = (u32x4*)Lih0;
    for (int i = tid; i < NW * II / 8; i += THREADS) d0[i] = s0[i];
    const u32x4* s1 = (const u32x4*)g1s; u32x4* d1 = (u32x4*)Lhh0;
    for (int i = tid; i < NW * HH / 8; i += THREADS) d1[i] = s1[i];
    const u32x4* s2 = (const u32x4*)g2s; u32x4* d2 = (u32x4*)Lih1;
    for (int i = tid; i < NW * HH / 8; i += THREADS) d2[i] = s2[i];
    const u32x4* s3 = (const u32x4*)g3s; u32x4* d3 = (u32x4*)Lhh1;
    for (int i = tid; i < NW * HH / 8; i += THREADS) d3[i] = s3[i];
  }
#endif
  __syncthreads();

  // ---- lane / wave mapping (wave32) ----
  const int lane  = tid & 31;
  const int wave  = tid >> 5;         // 0..7
  const int half  = lane >> 4;        // K-half selector for fragments
  const int lrow  = lane & 15;
  const int mtile = wave >> 2;        // 0..1  (batch 0..15 / 16..31)
  const int ntile = wave & 3;         // 0..3  (16-col groups within slab)
  const int ma    = mtile * 16 + lrow;        // A-fragment row (batch)
  const int nl    = ntile * 16 + lrow;        // B-fragment row in LDS slab
  const int nc    = n0 + nl;                  // global output column

  const float bs0 = b0g[nc];
  const float bs1 = b1g[nc];

  unsigned bar_gen = 0;
  int p = 0;                                   // parity: h*buf[p] holds h_{t-1}

  auto gbar = [&]() {
    ++bar_gen;
    __builtin_amdgcn_fence(__ATOMIC_RELEASE, "agent");
    __syncthreads();
    if (tid == 0) {
      __hip_atomic_fetch_add(barcnt, 1u, __ATOMIC_RELAXED, __HIP_MEMORY_SCOPE_AGENT);
      const unsigned tgt = (unsigned)NWG * bar_gen;
      while (__hip_atomic_load(barcnt, __ATOMIC_RELAXED, __HIP_MEMORY_SCOPE_AGENT) < tgt)
        __builtin_amdgcn_s_sleep(2);
    }
    __syncthreads();
    __builtin_amdgcn_fence(__ATOMIC_ACQUIRE, "agent");
  };

  for (int t = 0; t < TT; ++t) {
    // prefetch next timestep's x rows into the cache hierarchy (global_prefetch)
    if (t + 1 < TT) {
      const __bf16* nxt = xb + ((size_t)ma * TT + (t + 1)) * II;
      __builtin_prefetch(nxt, 0, 1);
      __builtin_prefetch(nxt + 128, 0, 1);
    }

    // ================= layer 0:  h0' = tanh(x_t @ Wih0 + h0 @ Whh0 + b0) ====
    v8f acc = {};
    {
      const __bf16* arow = xb + ((size_t)ma * TT + t) * II;
      const __bf16* brow = Lih0 + (size_t)nl * II;
      #pragma unroll 4
      for (int kt = 0; kt < II / 32; ++kt)
        acc = WMMA_BF16(load_fragA(arow + kt * 32, half),
                        load_fragB(brow + kt * 32, half), acc);

      const __bf16* hrow  = h0buf + (size_t)p * BB * HH + (size_t)ma * HH;
      const __bf16* brow2 = Lhh0 + (size_t)nl * HH;
      #pragma unroll 4
      for (int kt = 0; kt < HH / 32; ++kt)
        acc = WMMA_BF16(load_fragA(hrow + kt * 32, half),
                        load_fragB(brow2 + kt * 32, half), acc);
    }
    {
      __bf16* hnew = h0buf + (size_t)(1 - p) * BB * HH;
      #pragma unroll
      for (int r = 0; r < 8; ++r) {
        float v = fast_tanh(acc[r] + bs0);      // D layout: M = mtile*16 + half*8 + r, N = lrow
        int mr = mtile * 16 + half * 8 + r;
        hnew[(size_t)mr * HH + nc] = (__bf16)v;
        if (t == TT - 1)
          out[(size_t)BB * TT * HH + (size_t)mr * HH + nc] = v;  // hT layer 0
      }
    }
    gbar();   // all slabs of h0' visible device-wide

    // ================= layer 1:  h1' = tanh(h0' @ Wih1 + h1 @ Whh1 + b1) ====
    acc = (v8f){};
    {
      const __bf16* arow  = h0buf + (size_t)(1 - p) * BB * HH + (size_t)ma * HH;
      const __bf16* brow  = Lih1 + (size_t)nl * HH;
      #pragma unroll 4
      for (int kt = 0; kt < HH / 32; ++kt)
        acc = WMMA_BF16(load_fragA(arow + kt * 32, half),
                        load_fragB(brow + kt * 32, half), acc);

      const __bf16* hrow  = h1buf + (size_t)p * BB * HH + (size_t)ma * HH;
      const __bf16* brow2 = Lhh1 + (size_t)nl * HH;
      #pragma unroll 4
      for (int kt = 0; kt < HH / 32; ++kt)
        acc = WMMA_BF16(load_fragA(hrow + kt * 32, half),
                        load_fragB(brow2 + kt * 32, half), acc);
    }
    {
      __bf16* hnew = h1buf + (size_t)(1 - p) * BB * HH;
      #pragma unroll
      for (int r = 0; r < 8; ++r) {
        float v = fast_tanh(acc[r] + bs1);
        int mr = mtile * 16 + half * 8 + r;
        hnew[(size_t)mr * HH + nc] = (__bf16)v;
        // streamed 128MB output: keep it out of the working set (TH_NT)
        __builtin_nontemporal_store(v, &out[((size_t)mr * TT + t) * HH + nc]);
        if (t == TT - 1)
          out[(size_t)BB * TT * HH + (size_t)BB * HH + (size_t)mr * HH + nc] = v; // hT layer 1
      }
    }
    gbar();   // h1' visible; all reads of parity-p buffers done
    p ^= 1;
  }
}

// ---------------------------------------------------------------------------
// Prep kernels
// ---------------------------------------------------------------------------
__global__ void cvt_bf16(const float* __restrict__ s, __bf16* __restrict__ d, size_t n) {
  size_t i = (size_t)blockIdx.x * blockDim.x + threadIdx.x;
  size_t stride = (size_t)gridDim.x * blockDim.x;
  for (; i < n; i += stride) d[i] = (__bf16)s[i];
}

// src: [K][N] fp32 row-major  ->  dst: [N][K] bf16 row-major (transposed)
__global__ void cvt_transpose(const float* __restrict__ src, __bf16* __restrict__ dst,
                              int K, int N) {
  int idx = blockIdx.x * blockDim.x + threadIdx.x;
  if (idx < K * N) {
    int k = idx / N, n = idx % N;
    dst[(size_t)n * K + k] = (__bf16)src[idx];
  }
}

__global__ void init_state(__bf16* h0, __bf16* h1, unsigned* cnt) {
  int i = blockIdx.x * blockDim.x + threadIdx.x;
  if (i < 2 * BB * HH) { h0[i] = (__bf16)0.0f; h1[i] = (__bf16)0.0f; }
  if (i == 0) *cnt = 0u;
}

// ---------------------------------------------------------------------------
extern "C" void kernel_launch(void* const* d_in, const int* in_sizes, int n_in,
                              void* d_out, int out_size, void* d_ws, size_t ws_size,
                              hipStream_t stream) {
  (void)in_sizes; (void)n_in; (void)out_size; (void)ws_size;

  const float* x    = (const float*)d_in[0];
  const float* Wih0 = (const float*)d_in[1];
  const float* Whh0 = (const float*)d_in[2];
  const float* b0   = (const float*)d_in[3];
  const float* Wih1 = (const float*)d_in[4];
  const float* Whh1 = (const float*)d_in[5];
  const float* b1   = (const float*)d_in[6];
  float* out = (float*)d_out;

  // workspace layout (~35.6 MB): x_bf16 | WT (4 mats) | h0[2] | h1[2] | barrier
  char* ws = (char*)d_ws;
  size_t off = 0;
  __bf16* xb = (__bf16*)(ws + off); off += (size_t)BB * TT * II * sizeof(__bf16);
  __bf16* wt = (__bf16*)(ws + off); off += ((size_t)HH * II + 3 * (size_t)HH * HH) * sizeof(__bf16);
  __bf16* h0buf = (__bf16*)(ws + off); off += (size_t)2 * BB * HH * sizeof(__bf16);
  __bf16* h1buf = (__bf16*)(ws + off); off += (size_t)2 * BB * HH * sizeof(__bf16);
  off = (off + 255) & ~(size_t)255;
  unsigned* barcnt = (unsigned*)(ws + off);

  __bf16* wt_ih0 = wt;
  __bf16* wt_hh0 = wt_ih0 + (size_t)HH * II;
  __bf16* wt_ih1 = wt_hh0 + (size_t)HH * HH;
  __bf16* wt_hh1 = wt_ih1 + (size_t)HH * HH;

  // 1) convert x to bf16 (same [B][T][I] layout -> contiguous A-fragment rows)
  cvt_bf16<<<4096, 256, 0, stream>>>(x, xb, (size_t)BB * TT * II);
  // 2) transpose+convert weights to N-major bf16 slabs
  cvt_transpose<<<(II * HH + 255) / 256, 256, 0, stream>>>(Wih0, wt_ih0, II, HH);
  cvt_transpose<<<(HH * HH + 255) / 256, 256, 0, stream>>>(Whh0, wt_hh0, HH, HH);
  cvt_transpose<<<(HH * HH + 255) / 256, 256, 0, stream>>>(Wih1, wt_ih1, HH, HH);
  cvt_transpose<<<(HH * HH + 255) / 256, 256, 0, stream>>>(Whh1, wt_hh1, HH, HH);
  // 3) zero hidden state (parity 0) and barrier counter
  init_state<<<(2 * BB * HH + 255) / 256, 256, 0, stream>>>(h0buf, h1buf, barcnt);

  // 4) persistent time loop: 8 WGs, 224 KB LDS each (CDNA5: 320 KB/WGP)
  (void)hipFuncSetAttribute((const void*)rnn_persistent,
                            hipFuncAttributeMaxDynamicSharedMemorySize,
                            (int)SMEM_BYTES);
  rnn_persistent<<<NWG, THREADS, SMEM_BYTES, stream>>>(xb, wt, b0, b1,
                                                       h0buf, h1buf, out, barcnt);
}
